// MixedLinearV2_21663815041824
// MI455X (gfx1250) — matrix operand order repaired and checked by reading, not compiled, not run on hardware
//
#include <hip/hip_runtime.h>
#include <hip/hip_bf16.h>

// ---------------------------------------------------------------------------
// MixedLinearV2 as a single split-bf16 WMMA GEMM (gfx1250 / CDNA5, wave32).
//   y[m,n] = sum_k x[m,k] * W[n,k] * coeff(n,k) + b[n] * rb(n)
// M = 8*4096 = 32768, K = 1024, N = 4096, fp32 in/out.
// Compute: v_wmma_f32_16x16x32_bf16, 3-term split (hi*hi + hi*lo + lo*hi)
// for near-fp32 accuracy at 3x bf16-rate (vs 8x cost of f32 16x16x4 WMMA).
// ---------------------------------------------------------------------------

#define M_TOTAL 32768
#define N_TOTAL 4096
#define K_TOTAL 1024

#define BM 128
#define BN 64
#define BK 32
#define KSTEPS (K_TOTAL / BK)   // 32
#define LDA 40                  // 32 halves + 8 pad  -> 80B row stride
#define LDB 40

typedef __attribute__((ext_vector_type(16))) __bf16 v16bf;
typedef __attribute__((ext_vector_type(8)))  float  v8f;

static __device__ inline v8f wmma_bf16(v16bf a, v16bf b, v8f c) {
    // 8 args: (neg_a, A, neg_b, B, c_mod, C, reuse_a, reuse_b)
    return __builtin_amdgcn_wmma_f32_16x16x32_bf16(false, a, false, b,
                                                   (short)0, c, false, false);
}

// Load a 16-half fragment as two 16-byte LDS reads (matches ISA VGPR packing:
// consecutive halves -> consecutive dword lanes).
static __device__ inline v16bf ld_frag(const __bf16* p0, const __bf16* p1) {
    union { uint4 q[2]; v16bf v; } u;
    u.q[0] = *(const uint4*)p0;
    u.q[1] = *(const uint4*)p1;
    return u.v;
}

static __device__ inline void split1(float v, __bf16* ph, __bf16* pl) {
    __bf16 h = (__bf16)v;            // RNE high part
    *ph = h;
    *pl = (__bf16)(v - (float)h);    // residual low part
}

static __device__ inline void split4(float4 v, float c, __bf16* ph, __bf16* pl) {
    split1(v.x * c, ph + 0, pl + 0);
    split1(v.y * c, ph + 1, pl + 1);
    split1(v.z * c, ph + 2, pl + 2);
    split1(v.w * c, ph + 3, pl + 3);
}

__global__ __launch_bounds__(256)
void mixed_linear_wmma_kernel(const float* __restrict__ x,
                              const float* __restrict__ wts,   // [9] mixing weights
                              const float* __restrict__ W,     // [4096,1024]
                              const float* __restrict__ bias,  // [4096]
                              float* __restrict__ out) {
    __shared__ alignas(16) __bf16 Ah[2][BM * LDA];
    __shared__ alignas(16) __bf16 Al[2][BM * LDA];
    __shared__ alignas(16) __bf16 Bh[2][BN * LDB];
    __shared__ alignas(16) __bf16 Bl[2][BN * LDB];

    const int tid   = threadIdx.x;
    const int lane  = tid & 31;
    const int wave  = tid >> 5;        // 0..7
    const int waveM = wave & 3;        // 4 waves along M (32 rows each)
    const int waveN = wave >> 2;       // 2 waves along N (32 cols each)
    const int lmod  = lane & 15;
    const int lhalf = lane >> 4;

    const int tileM = blockIdx.y * BM;
    const int tileN = blockIdx.x * BN;

    // Mixing weights (uniform -> scalar loads).
    const float w0 = wts[0], w1 = wts[1], w2 = wts[2];
    const float w3 = wts[3], w4 = wts[4], w5 = wts[5];
    const float w6 = wts[6], w7 = wts[7], w8 = wts[8];

    // ---- loader roles -----------------------------------------------------
    // A tile: 128 rows x 32 f32 -> 16 f32/thread (4 x float4)
    const int rA = tid >> 1;                 // 0..127
    const int cA = (tid & 1) * 16;           // 0 or 16
    const float* gA = x + (size_t)(tileM + rA) * K_TOTAL + cA;
    // B tile: 64 rows x 32 f32 -> 8 f32/thread (2 x float4)
    const int rB = tid >> 2;                 // 0..63
    const int cB = (tid & 3) * 8;            // 0,8,16,24
    const float* gB = W + (size_t)(tileN + rB) * K_TOTAL + cB;

    // Per-row mixing sums for this thread's B row (o fixed for whole K loop).
    const int oB = tileN + rB;
    const float s0 = (oB < 1024 ? w0 : 0.f) + (oB < 1536 ? w1 : 0.f) + (oB < 2048 ? w2 : 0.f);
    const float s1 = (oB < 1536 ? w3 : 0.f) + (oB < 2304 ? w4 : 0.f) + (oB < 3072 ? w5 : 0.f);
    const float s2 = (oB < 2048 ? w6 : 0.f) + (oB < 3072 ? w7 : 0.f) + w8;

    float4 ra[4];
    float4 rb[2];

    auto gloadA = [&](int kt) {
        const float* p = gA + kt * BK;
#pragma unroll
        for (int j = 0; j < 4; ++j) ra[j] = *(const float4*)(p + j * 4);
    };
    auto gloadB = [&](int kt) {
        const float* p = gB + kt * BK;
#pragma unroll
        for (int j = 0; j < 2; ++j) rb[j] = *(const float4*)(p + j * 4);
    };
    auto stA = [&](int bb) {
        __bf16* dh = &Ah[bb][rA * LDA + cA];
        __bf16* dl = &Al[bb][rA * LDA + cA];
#pragma unroll
        for (int j = 0; j < 4; ++j) split4(ra[j], 1.0f, dh + j * 4, dl + j * 4);
    };
    auto stB = [&](int bb, int kt) {
        // K thresholds (512,768) are multiples of BK -> coeff uniform per k-tile.
        const float coeff = s2 + (kt < 24 ? s1 : 0.f) + (kt < 16 ? s0 : 0.f);
        __bf16* dh = &Bh[bb][rB * LDB + cB];
        __bf16* dl = &Bl[bb][rB * LDB + cB];
#pragma unroll
        for (int j = 0; j < 2; ++j) split4(rb[j], coeff, dh + j * 4, dl + j * 4);
    };

    // ---- prologue ---------------------------------------------------------
    gloadA(0); gloadB(0);
    stA(0);    stB(0, 0);
    gloadA(1); gloadB(1);
    __syncthreads();

    v8f acc[2][2];
#pragma unroll
    for (int mi = 0; mi < 2; ++mi)
#pragma unroll
        for (int ni = 0; ni < 2; ++ni) acc[mi][ni] = (v8f){0.f,0.f,0.f,0.f,0.f,0.f,0.f,0.f};

    // ---- main K loop: double-buffered LDS, global loads 2 tiles ahead -----
    for (int kt = 0; kt < KSTEPS; ++kt) {
        const int cur = kt & 1;
        if (kt + 1 < KSTEPS) { stA(cur ^ 1); stB(cur ^ 1, kt + 1); }
        if (kt + 2 < KSTEPS) { gloadA(kt + 2); gloadB(kt + 2); }
        if (kt + 4 < KSTEPS) {
            __builtin_prefetch(gA + (kt + 4) * BK, 0, 0);   // global_prefetch_b8
            __builtin_prefetch(gB + (kt + 4) * BK, 0, 0);
        }

        // B fragments: lane holds n = lmod, K chunks at c*8 and 16+c*8 (c=lane/16).
        v16bf fbh[2], fbl[2];
#pragma unroll
        for (int ni = 0; ni < 2; ++ni) {
            const int bo = (waveN * 32 + ni * 16 + lmod) * LDB + lhalf * 8;
            fbh[ni] = ld_frag(&Bh[cur][bo], &Bh[cur][bo + 16]);
            fbl[ni] = ld_frag(&Bl[cur][bo], &Bl[cur][bo + 16]);
        }
        // A fragments: lane holds m = lmod, K chunks at c*8 and 16+c*8 (c=lane/16).
#pragma unroll
        for (int mi = 0; mi < 2; ++mi) {
            const int ao = (waveM * 32 + mi * 16 + lmod) * LDA + lhalf * 8;
            const v16bf fah = ld_frag(&Ah[cur][ao], &Ah[cur][ao + 16]);
            const v16bf fal = ld_frag(&Al[cur][ao], &Al[cur][ao + 16]);
#pragma unroll
            for (int ni = 0; ni < 2; ++ni) {
                acc[mi][ni] = wmma_bf16(fah, fbh[ni], acc[mi][ni]);  // hi*hi
                acc[mi][ni] = wmma_bf16(fah, fbl[ni], acc[mi][ni]);  // hi*lo
                acc[mi][ni] = wmma_bf16(fal, fbh[ni], acc[mi][ni]);  // lo*hi
            }
        }
        __syncthreads();
    }

    // ---- epilogue: add mixed bias, store fp32 ------------------------------
    // C/D layout: VGPR v -> M = (lane/16)*8 + v, N = lane%16.
#pragma unroll
    for (int ni = 0; ni < 2; ++ni) {
        const int n = tileN + waveN * 32 + ni * 16 + lmod;
        const float t0 = (n < 1024 ? w0 : 0.f) + (n < 1536 ? w1 : 0.f) + (n < 2048 ? w2 : 0.f);
        const float t1 = (n < 1536 ? w3 : 0.f) + (n < 2304 ? w4 : 0.f) + (n < 3072 ? w5 : 0.f);
        const float t2 = (n < 2048 ? w6 : 0.f) + (n < 3072 ? w7 : 0.f) + w8;
        const float bm = bias[n] * (t0 + t1 + t2);
#pragma unroll
        for (int mi = 0; mi < 2; ++mi) {
            const int m0 = tileM + waveM * 32 + mi * 16 + lhalf * 8;
            float* op = out + (size_t)m0 * N_TOTAL + n;
#pragma unroll
            for (int v = 0; v < 8; ++v)
                op[(size_t)v * N_TOTAL] = acc[mi][ni][v] + bm;
        }
    }
}

extern "C" void kernel_launch(void* const* d_in, const int* in_sizes, int n_in,
                              void* d_out, int out_size, void* d_ws, size_t ws_size,
                              hipStream_t stream) {
    (void)in_sizes; (void)n_in; (void)out_size; (void)d_ws; (void)ws_size;
    const float* x    = (const float*)d_in[0];  // [8,4096,1024]
    const float* wts  = (const float*)d_in[1];  // [9]
    const float* W    = (const float*)d_in[2];  // [4096,1024]
    const float* bias = (const float*)d_in[3];  // [4096]
    float* out = (float*)d_out;                 // [8,4096,4096]

    dim3 grid(N_TOTAL / BN, M_TOTAL / BM);      // (64, 256)
    dim3 block(256);                            // 8 wave32s
    mixed_linear_wmma_kernel<<<grid, block, 0, stream>>>(x, wts, W, bias, out);
}
